// NerfMLP_41188736368853
// MI455X (gfx1250) — matrix operand order repaired and checked by reading, not compiled, not run on hardware
//
#include <hip/hip_runtime.h>
#include <cstdint>
#include <cstddef>

// ---------------------------------------------------------------------------
// Fully-fused NeRF MLP for gfx1250 (MI455X), wave32.
//   enc(2->40, pad 64) -> [40x256] relu -> 6x [256x256] relu -> [256x3] tanh/100
// v_wmma_f32_16x16x32_f16 for all dense layers. Weights stream L2 -> LDS with
// global_load_async_to_lds_b128 (ASYNCcnt), double-buffered and software-
// pipelined so copies overlap compute; 4 waves share each staged tile.
// Dual accumulator chains per stage give 2 independent WMMA streams.
// ---------------------------------------------------------------------------

typedef __attribute__((ext_vector_type(16))) _Float16 v16h;
typedef __attribute__((ext_vector_type(8)))  _Float16 v8h;
typedef __attribute__((ext_vector_type(8)))  float    v8f;

#define HIDDEN          256
#define NL_PER_STAGE    4           // column tiles staged into LDS at once
#define NSTAGES         4           // 16 column tiles / 4 per stage
#define KCH_HID         8           // 256/32 K-chunks
#define KCH_IN          2           // 64/32 K-chunks (enc padded 40 -> 64)
#define ACT_STRIDE      264         // halves/row: 16B-aligned rows, bank-skewed
#define WAVES_PER_BLOCK 4
#define THREADS_PER_BLK (WAVES_PER_BLOCK * 32)
#define PTS_PER_WAVE    16
#define PTS_PER_BLOCK   (WAVES_PER_BLOCK * PTS_PER_WAVE)

// One 32x16 B-tile = 32 lanes * 16 halves = 512 halves; lane L's 16 halves
// (K = h + 16*(L/16), col = 16n + L%16) are contiguous (32B per lane).
// Layer tile order: [stage][kchunk][ntile_local] -> each stage is one
// contiguous region, so staging is a flat async copy.
#define TILE_HALVES     512
#define TILES_IN        (NSTAGES * KCH_IN  * NL_PER_STAGE)       // 32
#define TILES_HID_L     (NSTAGES * KCH_HID * NL_PER_STAGE)       // 128
#define TILES_TOTAL     (TILES_IN + 6 * TILES_HID_L)             // 800
#define PACKED_HALVES   (TILES_TOTAL * TILE_HALVES)              // 409600

#define WSTAGE_HALVES   (KCH_HID * NL_PER_STAGE * TILE_HALVES)   // 16384 (32KB)
#define ACT_HALVES      (16 * ACT_STRIDE)                        // per wave
#define SMEM_BYTES      ((2 * WSTAGE_HALVES + WAVES_PER_BLOCK * ACT_HALVES) * 2)

// ---------------------------------------------------------------------------
// Prep kernel: fp32 weights -> packed f16 B-fragment layout in workspace.
// ---------------------------------------------------------------------------
__global__ void pack_weights_kernel(const float* __restrict__ W_in,
                                    const float* __restrict__ W_hid,
                                    _Float16* __restrict__ wpk) {
    int tid = blockIdx.x * 256 + threadIdx.x;
    if (tid >= PACKED_HALVES) return;
    int tile   = tid >> 9;                 // /512
    int within = tid & 511;
    int lane   = within >> 4;
    int h      = within & 15;
    int kk     = h + ((lane >> 4) << 4);   // 0..31 within K-chunk
    int colL   = lane & 15;

    float val;
    if (tile < TILES_IN) {                  // input layer (K padded to 64)
        int s  = tile >> 3;                 // 8 tiles per stage (2 kch * 4 nl)
        int r  = tile & 7;
        int c  = r >> 2, nl = r & 3;
        int n  = s * NL_PER_STAGE + nl;
        int K  = c * 32 + kk;
        int col = n * 16 + colL;
        val = (K < 40) ? W_in[K * HIDDEN + col] : 0.0f;
    } else {                                // hidden layers
        int t  = tile - TILES_IN;
        int l  = t >> 7;                    // /128
        int r  = t & 127;
        int s  = r >> 5;                    // 32 tiles per stage (8 kch * 4 nl)
        int r2 = r & 31;
        int c  = r2 >> 2, nl = r2 & 3;
        int n  = s * NL_PER_STAGE + nl;
        int K  = c * 32 + kk;
        int col = n * 16 + colL;
        val = W_hid[((l * HIDDEN + K) << 8) + col];
    }
    wpk[tid] = (_Float16)val;
}

// ---------------------------------------------------------------------------
// CDNA5 async global->LDS copy (ASYNCcnt-tracked), flat 16B granules.
// Every thread issues bytes/(THREADS*16) instructions -> per-wave ASYNCcnt
// increments are uniform and known at compile time.
// ---------------------------------------------------------------------------
__device__ __forceinline__ void async_copy_stage(uint32_t lds_base_bytes,
                                                 const _Float16* gsrc,
                                                 int bytes, int tid) {
    for (int off = tid * 16; off < bytes; off += THREADS_PER_BLK * 16) {
        asm volatile("global_load_async_to_lds_b128 %0, %1, %2 offset:0"
                     :
                     : "v"(lds_base_bytes + (uint32_t)off), "v"(off), "s"(gsrc)
                     : "memory");
    }
}

// ---------------------------------------------------------------------------
// Positional encoding value for padded column d (0..63).
// ---------------------------------------------------------------------------
__device__ __forceinline__ float encval(int d, float x0, float x1) {
    if (d >= 40) return 0.0f;
    const float xi = (d < 20) ? x0 : x1;
    const int   r  = (d < 20) ? d : d - 20;
    const int   e  = (r < 10) ? r : r - 10;
    const float a  = xi * (float)(1 << e);
    return (r < 10) ? __sinf(a) : __cosf(a);
}

// ---------------------------------------------------------------------------
// One dense layer, double-buffered LDS weight staging + dual WMMA chains.
// ---------------------------------------------------------------------------
template <int KCH>
__device__ __forceinline__ void mlp_layer_staged(_Float16* __restrict__ A,
                                                 _Float16* __restrict__ wbuf,
                                                 uint32_t wbuf_base_bytes,
                                                 const _Float16* __restrict__ gW,
                                                 const float* __restrict__ bias,
                                                 int lane, int tid) {
    const int row = lane & 15;
    const int g   = lane >> 4;
    const int stageHalves = KCH * NL_PER_STAGE * TILE_HALVES;
    const int stageBytes  = stageHalves * 2;
    constexpr int INFLIGHT = (KCH * NL_PER_STAGE * TILE_HALVES * 2)
                             / (THREADS_PER_BLK * 16);   // async ops per stage

    // Kick off stage 0 copy into buffer 0 (overlaps with A-fragment hoist).
    async_copy_stage(wbuf_base_bytes, gW, stageBytes, tid);

    // Hoist all A fragments (contiguous K order, same as the B packing):
    // lane<16: K {32c+0..15}; lane>=16: K {32c+16..31}
    v16h a[KCH];
#pragma unroll
    for (int c = 0; c < KCH; ++c) {
        const _Float16* p = &A[row * ACT_STRIDE + c * 32 + g * 16];
        v8h lo = *(const v8h*)p;
        v8h hi = *(const v8h*)(p + 8);
#pragma unroll
        for (int i = 0; i < 8; ++i) { a[c][i] = lo[i]; a[c][i + 8] = hi[i]; }
    }

    for (int s = 0; s < NSTAGES; ++s) {
        const int cur = s & 1;
        if (s + 1 < NSTAGES) {
            // Prefetch next stage into the other buffer, then wait for only
            // the *current* stage (async loads complete in order per wave).
            async_copy_stage(wbuf_base_bytes + (uint32_t)((cur ^ 1) * WSTAGE_HALVES * 2),
                             gW + (size_t)(s + 1) * stageHalves, stageBytes, tid);
            asm volatile("s_wait_asynccnt %0" :: "i"(INFLIGHT) : "memory");
        } else {
            asm volatile("s_wait_asynccnt 0" ::: "memory");
        }
        __syncthreads();                       // staged tile visible to block

        _Float16* wb = wbuf + (size_t)cur * WSTAGE_HALVES;
#pragma unroll
        for (int p = 0; p < NL_PER_STAGE / 2; ++p) {
            const int nl0 = 2 * p;
            const int n0  = s * NL_PER_STAGE + nl0;
            const float b0 = bias[n0 * 16 + row];        // col = 16n + lane%16
            const float b1 = bias[(n0 + 1) * 16 + row];
            v8f acc0 = {}, acc1 = {};
#pragma unroll
            for (int c = 0; c < KCH; ++c) {
                v16h bf0 = *(const v16h*)(wb
                             + (size_t)(c * NL_PER_STAGE + nl0) * TILE_HALVES
                             + lane * 16);
                v16h bf1 = *(const v16h*)(wb
                             + (size_t)(c * NL_PER_STAGE + nl0 + 1) * TILE_HALVES
                             + lane * 16);
                acc0 = __builtin_amdgcn_wmma_f32_16x16x32_f16(
                           false, a[c], false, bf0, (short)0, acc0, false, false);
                acc1 = __builtin_amdgcn_wmma_f32_16x16x32_f16(
                           false, a[c], false, bf1, (short)0, acc1, false, false);
            }
            // C layout: VGPR r -> M = r + 8*g, N = lane%16. Bias+ReLU -> f16.
#pragma unroll
            for (int r = 0; r < 8; ++r) {
                float v0 = acc0[r] + b0;
                float v1 = acc1[r] + b1;
                v0 = v0 > 0.0f ? v0 : 0.0f;
                v1 = v1 > 0.0f ? v1 : 0.0f;
                A[(r + g * 8) * ACT_STRIDE + n0 * 16 + row]       = (_Float16)v0;
                A[(r + g * 8) * ACT_STRIDE + (n0 + 1) * 16 + row] = (_Float16)v1;
            }
        }
        __syncthreads();   // block done reading this buffer before reuse
    }
}

// ---------------------------------------------------------------------------
// Main fused kernel: 4 waves/block, 16 points/wave.
// ---------------------------------------------------------------------------
__global__ __launch_bounds__(THREADS_PER_BLK)
void nerf_mlp_fused(const float* __restrict__ x,
                    const _Float16* __restrict__ wpk,
                    const float* __restrict__ b_in,
                    const float* __restrict__ b_hid,
                    const float* __restrict__ W_out,
                    const float* __restrict__ b_out,
                    float* __restrict__ out,
                    int npts) {
    extern __shared__ _Float16 smem[];
    _Float16* wbuf = smem;                                   // 2 x 32KB stages
    const uint32_t wbuf_base = (uint32_t)(uintptr_t)(void*)wbuf;

    const int tid  = threadIdx.x;
    const int lane = tid & 31;
    const int wave = tid >> 5;
    const int row  = lane & 15;
    const int g    = lane >> 4;
    _Float16* A = smem + 2 * WSTAGE_HALVES + wave * ACT_HALVES;

    const int basePt = blockIdx.x * PTS_PER_BLOCK + wave * PTS_PER_WAVE;

    // ---- positional encoding -> LDS (rows = points, cols 0..63, pad 0) ----
    {
        int pt = basePt + row;
        if (pt >= npts) pt = npts - 1;
        const float x0 = x[2 * pt];
        const float x1 = x[2 * pt + 1];
#pragma unroll
        for (int j = 0; j < 32; j += 2) {     // lane covers cols [32g, 32g+31]
            const int c0 = g * 32 + j;
            union { _Float16 h[2]; uint32_t u; } p;
            p.h[0] = (_Float16)encval(c0, x0, x1);
            p.h[1] = (_Float16)encval(c0 + 1, x0, x1);
            *(uint32_t*)&A[row * ACT_STRIDE + c0] = p.u;
        }
    }

    // ---- layer 0: 64(padded) -> 256 ----
    mlp_layer_staged<KCH_IN>(A, wbuf, wbuf_base, wpk, b_in, lane, tid);

    // ---- layers 1..6: 256 -> 256 ----
    for (int l = 0; l < 6; ++l) {
        const _Float16* gW = wpk + (size_t)(TILES_IN + l * TILES_HID_L)
                                       * TILE_HALVES;
        mlp_layer_staged<KCH_HID>(A, wbuf, wbuf_base, gW,
                                  b_hid + l * HIDDEN, lane, tid);
    }

    // ---- output layer: 256 -> 3, tanh/100 (48 dot products per wave) ----
    for (int t = lane; t < PTS_PER_WAVE * 3; t += 32) {
        const int pt = t / 3, oc = t % 3;
        float s = b_out[oc];
        const _Float16* ar = &A[pt * ACT_STRIDE];
#pragma unroll 8
        for (int k = 0; k < HIDDEN; ++k)
            s += (float)ar[k] * W_out[k * 3 + oc];
        const int gp = basePt + pt;
        if (gp < npts) out[gp * 3 + oc] = tanhf(s) * 0.01f;
    }
}

// ---------------------------------------------------------------------------
// Host launcher. d_in order: x, W_in, b_in, W_hid, b_hid, W_out, b_out.
// Workspace: packed f16 weights, PACKED_HALVES*2 = 819,200 bytes.
// ---------------------------------------------------------------------------
extern "C" void kernel_launch(void* const* d_in, const int* in_sizes, int n_in,
                              void* d_out, int out_size, void* d_ws, size_t ws_size,
                              hipStream_t stream) {
    const float* x     = (const float*)d_in[0];
    const float* W_in  = (const float*)d_in[1];
    const float* b_in  = (const float*)d_in[2];
    const float* W_hid = (const float*)d_in[3];
    const float* b_hid = (const float*)d_in[4];
    const float* W_out = (const float*)d_in[5];
    const float* b_out = (const float*)d_in[6];
    float* out = (float*)d_out;
    _Float16* wpk = (_Float16*)d_ws;

    const int npts = in_sizes[0] / 2;   // (B, N, 2) flattened

    const int pack_threads = 256;
    const int pack_blocks  = (PACKED_HALVES + pack_threads - 1) / pack_threads;
    pack_weights_kernel<<<pack_blocks, pack_threads, 0, stream>>>(W_in, W_hid, wpk);

    const int blocks = (npts + PTS_PER_BLOCK - 1) / PTS_PER_BLOCK;
    nerf_mlp_fused<<<blocks, THREADS_PER_BLK, SMEM_BYTES, stream>>>(
        x, wpk, b_in, b_hid, W_out, b_out, out, npts);
}
